// Autoregressive2dJoints_33363305955828
// MI455X (gfx1250) — matrix operand order repaired and checked by reading, not compiled, not run on hardware
//
#include <hip/hip_runtime.h>

typedef __attribute__((ext_vector_type(16))) __bf16 v16bf;
typedef __attribute__((ext_vector_type(8)))  float  v8f;

#define IN_DIM   34
#define IN_PAD   64
#define ENC_DIM  128
#define RNN_DIM  64
#define GATE_DIM 256
#define DEC_PAD  48
#define SEQ      128
#define NSEED    10
#define NDEC     (SEQ - NSEED)   // 118
#define BATCH    8192
#define WPB      2               // waves per block -> 256 blocks for WGP coverage
#define BLOCK    (WPB * 32)

// LDS layout sizes (elements)
#define SZ_WIH  (GATE_DIM * ENC_DIM)   // 32768 bf16
#define SZ_WHH  (GATE_DIM * RNN_DIM)   // 16384 bf16
#define SZ_WENC (ENC_DIM * IN_PAD)     //  8192 bf16
#define SZ_WDEC (DEC_PAD * RNN_DIM)    //  3072 bf16
#define ACT_PER_WAVE (16*ENC_DIM + 16*IN_PAD + 2*16*RNN_DIM)  // 5120 bf16
#define SMEM_BYTES (2*(SZ_WIH + SZ_WHH + SZ_WENC + SZ_WDEC) \
                    + 4*(ENC_DIM + GATE_DIM + DEC_PAD)       \
                    + 2*(WPB * ACT_PER_WAVE))                // ~143 KB

__device__ __forceinline__ v8f wmma_bf16(v16bf a, v16bf b, v8f c) {
  return __builtin_amdgcn_wmma_f32_16x16x32_bf16(false, a, false, b, (short)0, c, false, false);
}

// A-matrix 16x32 bf16 fragment from row-major LDS buffer [16][stride].
// Contiguous k order used here: lane<16 -> M=lane, K = k0..k0+15;
//             lane>=16 -> M=lane-16, K = k0+16..k0+31.
__device__ __forceinline__ v16bf load_a_frag(const __bf16* __restrict__ buf, int stride,
                                             int k0, int lane) {
  const int m  = lane & 15;
  const int kb = k0 + ((lane & 16) ? 16 : 0);
  const __bf16* p = buf + m * stride + kb;
  v16bf a;
#pragma unroll
  for (int i = 0; i < 16; ++i) a[i] = p[i];
  return a;
}

// B-matrix 32x16 bf16 fragment from [N][K] LDS weight store.
// lane<16 -> N=n0+lane, K=k0..k0+15 ; lane>=16 -> N=n0+lane-16, K=k0+16..k0+31.
__device__ __forceinline__ v16bf load_b_frag(const __bf16* __restrict__ buf, int K,
                                             int n0, int k0, int lane) {
  const int n  = n0 + (lane & 15);
  const int kb = k0 + ((lane & 16) ? 16 : 0);
  const __bf16* p = buf + n * K + kb;
  v16bf b;
#pragma unroll
  for (int i = 0; i < 16; ++i) b[i] = p[i];
  return b;
}

// Broadcast a per-lane (per-column) scalar into a C fragment.
__device__ __forceinline__ v8f bcast8(float v) {
  v8f r;
#pragma unroll
  for (int i = 0; i < 8; ++i) r[i] = v;
  return r;
}

// Store f32 C fragment as bf16 into row-major [16][stride] LDS activation buffer.
__device__ __forceinline__ void store_c_bf16(__bf16* __restrict__ buf, int stride,
                                             int n0, int lane, v8f c) {
  const int m0 = (lane & 16) ? 8 : 0;
  const int n  = n0 + (lane & 15);
#pragma unroll
  for (int r = 0; r < 8; ++r) buf[(m0 + r) * stride + n] = (__bf16)c[r];
}

__device__ __forceinline__ float sigmoidf_(float x) { return 1.0f / (1.0f + __expf(-x)); }

// gates = sIn(16x128) @ Wih^T + sHin(16x64) @ Whh^T + bias ; LSTM cell update.
// bg[g*4+t] holds this lane's bias scalar for gate g, column tile t (preloaded).
__device__ __forceinline__ void lstm_step(const __bf16* __restrict__ sIn,
                                          const __bf16* __restrict__ sHin,
                                          __bf16* __restrict__ sHout,
                                          const __bf16* __restrict__ sWih,
                                          const __bf16* __restrict__ sWhh,
                                          const float* __restrict__ bg,
                                          v8f c_frag[4], int lane) {
#pragma unroll
  for (int t = 0; t < 4; ++t) {           // 16-col tile of each gate
    v8f gi = bcast8(bg[0 * 4 + t]);
    v8f gf = bcast8(bg[1 * 4 + t]);
    v8f gg = bcast8(bg[2 * 4 + t]);
    v8f go = bcast8(bg[3 * 4 + t]);
#pragma unroll
    for (int kk = 0; kk < 4; ++kk) {      // K over ENC_DIM=128
      v16bf a = load_a_frag(sIn, ENC_DIM, kk * 32, lane);
      gi = wmma_bf16(a, load_b_frag(sWih, ENC_DIM,       t * 16, kk * 32, lane), gi);
      gf = wmma_bf16(a, load_b_frag(sWih, ENC_DIM,  64 + t * 16, kk * 32, lane), gf);
      gg = wmma_bf16(a, load_b_frag(sWih, ENC_DIM, 128 + t * 16, kk * 32, lane), gg);
      go = wmma_bf16(a, load_b_frag(sWih, ENC_DIM, 192 + t * 16, kk * 32, lane), go);
    }
#pragma unroll
    for (int kk = 0; kk < 2; ++kk) {      // K over RNN_DIM=64
      v16bf a = load_a_frag(sHin, RNN_DIM, kk * 32, lane);
      gi = wmma_bf16(a, load_b_frag(sWhh, RNN_DIM,       t * 16, kk * 32, lane), gi);
      gf = wmma_bf16(a, load_b_frag(sWhh, RNN_DIM,  64 + t * 16, kk * 32, lane), gf);
      gg = wmma_bf16(a, load_b_frag(sWhh, RNN_DIM, 128 + t * 16, kk * 32, lane), gg);
      go = wmma_bf16(a, load_b_frag(sWhh, RNN_DIM, 192 + t * 16, kk * 32, lane), go);
    }
    const int m0 = (lane & 16) ? 8 : 0;
    const int n  = t * 16 + (lane & 15);
#pragma unroll
    for (int r = 0; r < 8; ++r) {
      const float iv = sigmoidf_(gi[r]);
      const float fv = sigmoidf_(gf[r]);
      const float gv = tanhf(gg[r]);
      const float ov = sigmoidf_(go[r]);
      const float cn = fv * c_frag[t][r] + iv * gv;
      c_frag[t][r] = cn;
      sHout[(m0 + r) * RNN_DIM + n] = (__bf16)(ov * tanhf(cn));
    }
  }
}

// sIn(16x128) = relu(sX(16x64,padded) @ Wenc^T + enc_b)
__device__ __forceinline__ void encoder_step(const __bf16* __restrict__ sX,
                                             __bf16* __restrict__ sIn,
                                             const __bf16* __restrict__ sWenc,
                                             const float* __restrict__ be, int lane) {
#pragma unroll
  for (int nt = 0; nt < 8; ++nt) {
    v8f acc = bcast8(be[nt]);
#pragma unroll
    for (int kk = 0; kk < 2; ++kk) {
      v16bf a = load_a_frag(sX, IN_PAD, kk * 32, lane);
      acc = wmma_bf16(a, load_b_frag(sWenc, IN_PAD, nt * 16, kk * 32, lane), acc);
    }
#pragma unroll
    for (int r = 0; r < 8; ++r) acc[r] = fmaxf(acc[r], 0.0f);
    store_c_bf16(sIn, ENC_DIM, nt * 16, lane, acc);
  }
}

// d(16x48) = sH(16x64) @ Wdec^T + dec_b   (cols 34..47 are exact zeros by padding)
__device__ __forceinline__ void decoder_step(const __bf16* __restrict__ sH,
                                             const __bf16* __restrict__ sWdec,
                                             const float* __restrict__ bd,
                                             v8f d[3], int lane) {
#pragma unroll
  for (int nt = 0; nt < 3; ++nt) {
    v8f acc = bcast8(bd[nt]);
#pragma unroll
    for (int kk = 0; kk < 2; ++kk) {
      v16bf a = load_a_frag(sH, RNN_DIM, kk * 32, lane);
      acc = wmma_bf16(a, load_b_frag(sWdec, RNN_DIM, nt * 16, kk * 32, lane), acc);
    }
    d[nt] = acc;
  }
}

__device__ __forceinline__ void load_x_tile(const float* __restrict__ x, int bbase, int t,
                                            __bf16* __restrict__ sX, int lane) {
  for (int i = lane; i < 16 * IN_PAD; i += 32) {
    const int row = i >> 6;
    const int col = i & 63;
    float v = 0.0f;
    if (col < IN_DIM) v = x[(size_t)(bbase + row) * (SEQ * IN_DIM) + t * IN_DIM + col];
    sX[i] = (__bf16)v;
  }
}

__global__ __launch_bounds__(BLOCK) void lstm_ar_wmma_kernel(
    const float* __restrict__ x,     const float* __restrict__ enc_W,
    const float* __restrict__ enc_b, const float* __restrict__ w_ih,
    const float* __restrict__ w_hh,  const float* __restrict__ b_ih,
    const float* __restrict__ b_hh,  const float* __restrict__ dec_W,
    const float* __restrict__ dec_b, float* __restrict__ out) {
  extern __shared__ char smem_raw[];
  __bf16* sWih  = (__bf16*)smem_raw;                    // [256][128]
  __bf16* sWhh  = sWih  + SZ_WIH;                       // [256][64]
  __bf16* sWenc = sWhh  + SZ_WHH;                       // [128][64]  (K padded 34->64)
  __bf16* sWdec = sWenc + SZ_WENC;                      // [48][64]   (N padded 34->48)
  float*  sBenc = (float*)(sWdec + SZ_WDEC);            // [128]
  float*  sBgate = sBenc + ENC_DIM;                     // [256] = b_ih + b_hh
  float*  sBdec  = sBgate + GATE_DIM;                   // [48]
  __bf16* sAct   = (__bf16*)(sBdec + DEC_PAD);

  const int tid = threadIdx.x;
  // ---- per-block weight pack: fp32 -> bf16, [N][K] with zero padding ----
  for (int i = tid; i < SZ_WIH; i += BLOCK) sWih[i] = (__bf16)w_ih[i];
  for (int i = tid; i < SZ_WHH; i += BLOCK) sWhh[i] = (__bf16)w_hh[i];
  for (int i = tid; i < SZ_WENC; i += BLOCK) {
    const int n = i >> 6, k = i & 63;
    sWenc[i] = (k < IN_DIM) ? (__bf16)enc_W[n * IN_DIM + k] : (__bf16)0.0f;
  }
  for (int i = tid; i < SZ_WDEC; i += BLOCK) {
    const int n = i >> 6;
    sWdec[i] = (n < IN_DIM) ? (__bf16)dec_W[i] : (__bf16)0.0f;
  }
  for (int i = tid; i < ENC_DIM;  i += BLOCK) sBenc[i]  = enc_b[i];
  for (int i = tid; i < GATE_DIM; i += BLOCK) sBgate[i] = b_ih[i] + b_hh[i];
  for (int i = tid; i < DEC_PAD;  i += BLOCK) sBdec[i]  = (i < IN_DIM) ? dec_b[i] : 0.0f;
  __syncthreads();

  const int wave = tid >> 5;
  const int lane = tid & 31;
  __bf16* sIn = sAct + wave * ACT_PER_WAVE;             // [16][128] lstm input
  __bf16* sX  = sIn + 16 * ENC_DIM;                     // [16][64]  encoder input
  __bf16* sH0 = sX  + 16 * IN_PAD;                      // [16][64]  h ping
  __bf16* sH1 = sH0 + 16 * RNN_DIM;                     // [16][64]  h pong

  const int bbase = blockIdx.x * (WPB * 16) + wave * 16;

  for (int i = lane; i < 16 * RNN_DIM; i += 32) { sH0[i] = (__bf16)0.0f; sH1[i] = (__bf16)0.0f; }
  for (int i = lane; i < 16 * IN_PAD;  i += 32) sX[i] = (__bf16)0.0f;

  // ---- hoist per-lane bias scalars out of the time loop ----
  const int lc = lane & 15;
  float bg[16];                 // [gate][tile]
#pragma unroll
  for (int g = 0; g < 4; ++g)
#pragma unroll
    for (int t = 0; t < 4; ++t) bg[g * 4 + t] = sBgate[g * 64 + t * 16 + lc];
  float be[8];
#pragma unroll
  for (int nt = 0; nt < 8; ++nt) be[nt] = sBenc[nt * 16 + lc];
  float bd[3];
#pragma unroll
  for (int nt = 0; nt < 3; ++nt) bd[nt] = sBdec[nt * 16 + lc];

  v8f c_frag[4];
#pragma unroll
  for (int t = 0; t < 4; ++t)
#pragma unroll
    for (int r = 0; r < 8; ++r) c_frag[t][r] = 0.0f;

  __bf16* hc = sH0;
  __bf16* hn = sH1;

  // ---- seed phase: 10 encoder+LSTM steps ----
  for (int t = 0; t < NSEED; ++t) {
    load_x_tile(x, bbase, t, sX, lane);
    encoder_step(sX, sIn, sWenc, be, lane);
    lstm_step(sIn, hc, hn, sWih, sWhh, bg, c_frag, lane);
    { __bf16* tmp = hc; hc = hn; hn = tmp; }
  }

  // prev = x[:, NSEED-1, :] as C-layout fragments
  const int m0 = (lane & 16) ? 8 : 0;
  v8f p[3];
#pragma unroll
  for (int nt = 0; nt < 3; ++nt) {
    const int col = nt * 16 + lc;
#pragma unroll
    for (int r = 0; r < 8; ++r) {
      float v = 0.0f;
      if (col < IN_DIM)
        v = x[(size_t)(bbase + m0 + r) * (SEQ * IN_DIM) + (NSEED - 1) * IN_DIM + col];
      p[nt][r] = v;
    }
  }

  // dec_input = encoder(decoder(h))
  v8f d[3];
  decoder_step(hc, sWdec, bd, d, lane);
#pragma unroll
  for (int nt = 0; nt < 3; ++nt) store_c_bf16(sX, IN_PAD, nt * 16, lane, d[nt]);
  encoder_step(sX, sIn, sWenc, be, lane);

  // ---- decode phase: 118 autoregressive steps ----
  for (int s = 0; s < NDEC; ++s) {
    lstm_step(sIn, hc, hn, sWih, sWhh, bg, c_frag, lane);
    { __bf16* tmp = hc; hc = hn; hn = tmp; }
    decoder_step(hc, sWdec, bd, d, lane);
#pragma unroll
    for (int nt = 0; nt < 3; ++nt) {
      v8f f;
#pragma unroll
      for (int r = 0; r < 8; ++r) f[r] = d[nt][r] + p[nt][r];
      p[nt] = f;
      const int col = nt * 16 + lc;
      if (col < IN_DIM) {
#pragma unroll
        for (int r = 0; r < 8; ++r)
          out[(size_t)(bbase + m0 + r) * (NDEC * IN_DIM) + s * IN_DIM + col] = f[r];
      }
      store_c_bf16(sX, IN_PAD, nt * 16, lane, d[nt]);   // feedback (pre-residual)
    }
    encoder_step(sX, sIn, sWenc, be, lane);
  }
}

extern "C" void kernel_launch(void* const* d_in, const int* in_sizes, int n_in,
                              void* d_out, int out_size, void* d_ws, size_t ws_size,
                              hipStream_t stream) {
  const float* x     = (const float*)d_in[0];
  const float* enc_W = (const float*)d_in[1];
  const float* enc_b = (const float*)d_in[2];
  const float* w_ih  = (const float*)d_in[3];
  const float* w_hh  = (const float*)d_in[4];
  const float* b_ih  = (const float*)d_in[5];
  const float* b_hh  = (const float*)d_in[6];
  const float* dec_W = (const float*)d_in[7];
  const float* dec_b = (const float*)d_in[8];
  float* out = (float*)d_out;

  (void)in_sizes; (void)n_in; (void)out_size; (void)d_ws; (void)ws_size;

  const int blocks = BATCH / (WPB * 16);   // 256 blocks, 2 waves each, 16 batch rows/wave
  hipFuncSetAttribute((const void*)lstm_ar_wmma_kernel,
                      hipFuncAttributeMaxDynamicSharedMemorySize, (int)SMEM_BYTES);
  lstm_ar_wmma_kernel<<<blocks, BLOCK, SMEM_BYTES, stream>>>(
      x, enc_W, enc_b, w_ih, w_hh, b_ih, b_hh, dec_W, dec_b, out);
}
